// SpEncoderDecoderFHD_63591285785178
// MI455X (gfx1250) — matrix operand order repaired and checked by reading, not compiled, not run on hardware
//
#include <hip/hip_runtime.h>
#include <hip/hip_bf16.h>

// ---------------------------------------------------------------------------
// CDNA5 (gfx1250) masked sparse 3D U-Net.
// Every conv (subm / strided pool / transposed / 1x1 proj) is an implicit-GEMM
// on v_wmma_f32_16x16x32_f16 (wave32). Weights are pre-packed into WMMA
// B fragments (k = 16*(lane/16) + i); the A tile (im2col gather) is staged to LDS in 16-byte
// runs via gfx1250 global_load_async_to_lds_b128 (fallback: ld/st).
// ---------------------------------------------------------------------------

typedef _Float16 half_t;
typedef __attribute__((ext_vector_type(16))) _Float16 v16h;
typedef __attribute__((ext_vector_type(8)))  _Float16 v8h;
typedef __attribute__((ext_vector_type(8)))  float    v8f;
typedef __attribute__((ext_vector_type(4)))  int      v4i;

#define AS1 __attribute__((address_space(1)))
#define AS3 __attribute__((address_space(3)))

#define CONV_BLOCK 128   // 4 waves of 32

#if defined(__HIP_DEVICE_COMPILE__) && \
    __has_builtin(__builtin_amdgcn_global_load_async_to_lds_b128)
#define USE_ASYNC_LDS 1
#else
#define USE_ASYNC_LDS 0
#endif

__device__ __forceinline__ void wait_async_copies() {
#if USE_ASYNC_LDS
#if __has_builtin(__builtin_amdgcn_s_wait_asynccnt)
  __builtin_amdgcn_s_wait_asynccnt(0);
#else
  asm volatile("s_wait_asynccnt 0" ::: "memory");
#endif
#endif
}

struct ConvP {
  int ID, IH, IW;          // input spatial dims
  int OD, OH, OW;          // output spatial dims
  int SD, SH, SW;          // gather stride (1 for transposed convs)
  int PD, PH, PW;          // pad_lo (k-1-p for transposed convs)
  int Mtot;
  int has_bn, do_relu;
};

// ---------------------------------------------------------------------------
// Weight pre-pack: f32 weights (any of OIDHW / IODHW / OC layouts via element
// strides, optional tap flip) -> f16 WMMA B-fragments:
//   out[((kt*NT + n)*32 + lane)*16 + i],  K = (lane>>4)*16 + i, N = lane&15.
// ---------------------------------------------------------------------------
__global__ void pack_weights_kernel(const float* __restrict__ w,
                                    half_t* __restrict__ out,
                                    int Cin, int Cout, int KD, int KH, int KW,
                                    int flip, long wsCo, long wsCi,
                                    long wsKd, long wsKh, long wsKw,
                                    int totElems, int Ktot)
{
  const int e = blockIdx.x * blockDim.x + threadIdx.x;
  if (e >= totElems) return;
  const int i    = e & 15;
  const int lane = (e >> 4) & 31;
  const int tn   = e >> 9;            // kt*NT + n
  const int NTr  = Cout >> 4;
  const int kt   = tn / NTr;
  const int n    = tn - kt * NTr;
  const int kk   = ((lane >> 4) << 4) + i;
  const int kg   = kt * 32 + kk;
  const int nl   = lane & 15;
  const int co   = n * 16 + nl;
  half_t val = (half_t)0.0f;
  if (kg < Ktot) {
    const int tap = kg / Cin;
    const int ci  = kg - tap * Cin;
    const int KV  = KD * KH * KW;
    const int tw  = flip ? (KV - 1 - tap) : tap;   // flat flip == per-axis flip
    const int kd  = tw / (KH * KW);
    const int r2  = tw - kd * KH * KW;
    const int kh  = r2 / KW;
    const int kw  = r2 - kh * KW;
    val = (half_t)w[(long)co * wsCo + (long)ci * wsCi +
                    (long)kd * wsKd + (long)kh * wsKh + (long)kw * wsKw];
  }
  out[e] = val;
}

// ---------------------------------------------------------------------------
// Implicit-GEMM conv3d. Fully templated: WMMA loops static, tap decomposition
// and lhs-dilation (transposed conv) checks are compile-time.
//   NT = Cout/16 (n-tiles in registers), CL = log2(Cin),
//   LDC/LHC/LWC = lhs dilation per axis (1 for normal convs).
// ---------------------------------------------------------------------------
template <int NT, int KD, int KH, int KW, int CL, int LDC, int LHC, int LWC>
__global__ __launch_bounds__(CONV_BLOCK)
void conv3d_wmma_kernel(ConvP p,
    const half_t* __restrict__ in, const half_t* __restrict__ wpk,
    const float* __restrict__ bias,
    const float* __restrict__ bnb, const float* __restrict__ bng,
    const float* __restrict__ bnm, const float* __restrict__ bnv,
    const float* __restrict__ mask, const float* __restrict__ resid,
    float* __restrict__ outf, half_t* __restrict__ outh)
{
  constexpr int KHW    = KH * KW;
  constexpr int KV     = KD * KHW;
  constexpr int CIN    = 1 << CL;
  constexpr int CMASK  = CIN - 1;
  constexpr int KTOT   = KV * CIN;
  constexpr int KITERS = (KTOT + 31) / 32;
  constexpr int COUT   = NT * 16;

  // A-fragment staging buffers (double-buffered): [buf][wave][lane][16 halves]
  __shared__ __align__(32) half_t aFrag[2][4 * 512];

  const int tid    = threadIdx.x;
  const int wv     = tid >> 5;
  const int lane   = tid & 31;
  const int mblock = blockIdx.x * 64;

  // ---- staging role: each thread owns a fixed (kk8, ml) and 2 m-sites -----
  const int kk8 = tid & 3;            // which 8-half run of the 32-wide chunk
  const int ml  = (tid >> 2) & 15;    // local m row
  const int w4b = tid >> 6;           // first of two wave-tiles handled
  const int lg  = (kk8 >> 1) & 1;                // A lane group
  const int iib = (kk8 & 1) << 3;                // A half-index base (0 or 8)

  int oxA[2], oyA[2], ozA[2], obA[2], ldsoff[2];
  bool mvA[2];
#pragma unroll
  for (int j = 0; j < 2; ++j) {
    const int w4 = w4b + 2 * j;
    const int m  = mblock + w4 * 16 + ml;
    mvA[j] = (m < p.Mtot);
    int t = m;
    oxA[j] = t % p.OW; t /= p.OW;
    oyA[j] = t % p.OH; t /= p.OH;
    ozA[j] = t % p.OD; obA[j] = t / p.OD;
    ldsoff[j] = w4 * 512 + (((lg << 4) + ml) << 4) + iib;
  }

  v8f acc[NT];
#pragma unroll
  for (int n = 0; n < NT; ++n)
#pragma unroll
    for (int r = 0; r < 8; ++r) acc[n][r] = 0.0f;

  for (int kt = 0; kt < KITERS; ++kt) {
    const int  buf = kt & 1;
    const int  kg0 = kt * 32 + kk8 * 8;
    const bool kok = (kg0 < KTOT);
    const int  tap = kg0 >> CL;
    const int  ci0 = kg0 & CMASK;
    const int  kd  = tap / KHW;             // constant divides
    const int  r2  = tap - kd * KHW;
    const int  kh  = r2 / KW;
    const int  kw  = r2 - kh * KW;

    // -------- stage A: two 16-byte runs per thread -------------------------
#pragma unroll
    for (int j = 0; j < 2; ++j) {
      bool ok = kok && mvA[j];
      const int jz = ozA[j] * p.SD - p.PD + kd;
      const int jy = oyA[j] * p.SH - p.PH + kh;
      const int jx = oxA[j] * p.SW - p.PW + kw;
      int iz = jz, iy = jy, ix = jx;
      if constexpr (LDC != 1) {
        ok = ok && (jz >= 0) && ((jz & (LDC - 1)) == 0);
        iz = jz >> 1;   // LDC == 2
      }
      if constexpr (LHC != 1) {
        ok = ok && (jy >= 0) && ((jy & (LHC - 1)) == 0);
        iy = jy >> 1;
      }
      if constexpr (LWC != 1) {
        ok = ok && (jx >= 0) && ((jx & (LWC - 1)) == 0);
        ix = jx >> 1;
      }
      ok = ok && iz >= 0 && iz < p.ID && iy >= 0 && iy < p.IH &&
                 ix >= 0 && ix < p.IW;
      half_t* lp = &aFrag[buf][ldsoff[j]];
      if (ok) {
        const long gidx =
            ((long)((obA[j] * p.ID + iz) * p.IH + iy) * p.IW + ix) * CIN + ci0;
#if USE_ASYNC_LDS
        __builtin_amdgcn_global_load_async_to_lds_b128(
            (AS1 v4i*)(void*)(in + gidx),
            (AS3 v4i*)(void*)lp, 0, 0);
#else
        *reinterpret_cast<v8h*>(lp) = *reinterpret_cast<const v8h*>(in + gidx);
#endif
      } else {
        v8h z;
#pragma unroll
        for (int t8 = 0; t8 < 8; ++t8) z[t8] = (half_t)0.0f;
        *reinterpret_cast<v8h*>(lp) = z;
      }
    }
    wait_async_copies();
    __syncthreads();   // single barrier per chunk (A is double-buffered)

    // -------- WMMA: B fragments straight from global (pre-packed) ----------
    const v16h av =
        *reinterpret_cast<const v16h*>(&aFrag[buf][wv * 512 + (lane << 4)]);
#pragma unroll
    for (int n = 0; n < NT; ++n) {
      const v16h bv = *reinterpret_cast<const v16h*>(
          wpk + ((((long)kt * NT + n) * 32 + lane) << 4));
      acc[n] = __builtin_amdgcn_wmma_f32_16x16x32_f16(
          false, av, false, bv, (short)0, acc[n], false, false);
    }
  }

  // -------- fused epilogue: bias -> BN -> residual -> ReLU -> mask ---------
  // C/D layout: VGPR r holds M = r (+8 for lanes 16-31), N = lane&15.
  const int mrow = (lane >> 4) << 3;
  const int nl   = lane & 15;
#pragma unroll
  for (int n = 0; n < NT; ++n) {
    const int co = n * 16 + nl;
    float s = 1.0f, sh = 0.0f;
    if (p.has_bn) {
      s  = bng[co] * rsqrtf(bnv[co] + 1e-3f);
      sh = bnb[co] - bnm[co] * s;
    }
    const float bval = bias ? bias[co] : 0.0f;
#pragma unroll
    for (int r = 0; r < 8; ++r) {
      const int m = mblock + wv * 16 + mrow + r;
      if (m < p.Mtot) {
        float y = acc[n][r] + bval;
        if (p.has_bn) y = y * s + sh;
        if (resid)    y += resid[(long)m * COUT + co];
        if (p.do_relu) y = fmaxf(y, 0.0f);
        if (mask)     y *= mask[m];
        if (outf) outf[(long)m * COUT + co] = y;
        if (outh) outh[(long)m * COUT + co] = (half_t)y;
      }
    }
  }
}

// ---------------------------------------------------------------------------
// Support kernels
// ---------------------------------------------------------------------------
__global__ void scatter_kernel(const float* __restrict__ feats,
                               const int* __restrict__ coors,
                               float* __restrict__ dense,
                               float* __restrict__ mask, int N)
{
  const int v = blockIdx.x * blockDim.x + threadIdx.x;
  if (v >= N) return;
  const int b = coors[v * 4 + 0];
  const int z = coors[v * 4 + 1];
  const int y = coors[v * 4 + 2];
  const int x = coors[v * 4 + 3];
  const long site = ((long)(b * 41 + z) * 100 + y) * 88 + x;
  mask[site] = 1.0f;  // idempotent; races benign
#pragma unroll
  for (int c = 0; c < 16; ++c)
    atomicAdd(&dense[site * 16 + c], feats[(long)v * 16 + c]);
}

__global__ void cvt_f16_kernel(const float* __restrict__ src,
                               half_t* __restrict__ dst, long n)
{
  const long i = (long)blockIdx.x * blockDim.x + threadIdx.x;
  if (i < n) dst[i] = (half_t)src[i];
}

__global__ void maxpool_mask_kernel(const float* __restrict__ mi,
                                    float* __restrict__ mo,
                                    int ID, int IH, int IW,
                                    int OD, int OH, int OW,
                                    int KD, int KH, int KW,
                                    int SD, int SH, int SW,
                                    int PD, int PH, int PW)
{
  const long tot = 2L * OD * OH * OW;
  const long i = (long)blockIdx.x * blockDim.x + threadIdx.x;
  if (i >= tot) return;
  long t = i;
  const int ox = (int)(t % OW); t /= OW;
  const int oy = (int)(t % OH); t /= OH;
  const int oz = (int)(t % OD);
  const int b  = (int)(t / OD);
  float v = 0.0f;
  for (int kd = 0; kd < KD; ++kd) {
    const int iz = oz * SD - PD + kd;
    if (iz < 0 || iz >= ID) continue;
    for (int kh = 0; kh < KH; ++kh) {
      const int iy = oy * SH - PH + kh;
      if (iy < 0 || iy >= IH) continue;
      for (int kw = 0; kw < KW; ++kw) {
        const int ix = ox * SW - PW + kw;
        if (ix < 0 || ix >= IW) continue;
        v = fmaxf(v, mi[((long)(b * ID + iz) * IH + iy) * IW + ix]);
      }
    }
  }
  mo[i] = v;
}

__global__ void bev_out_kernel(const float* __restrict__ x4p,
                               float* __restrict__ out)
{
  const int C = 128, D = 2, H = 13, W = 11;
  const int i = blockIdx.x * blockDim.x + threadIdx.x;
  const int tot = 2 * C * D * H * W;
  if (i >= tot) return;
  int t = i;
  const int w  = t % W; t /= W;
  const int h  = t % H; t /= H;
  const int cd = t % (C * D);
  const int b  = t / (C * D);
  const int c = cd / D, d = cd % D;
  out[i] = x4p[(((long)(b * D + d) * H + h) * W + w) * C + c];
}

__global__ void u1_out_kernel(const float* __restrict__ u1,
                              float* __restrict__ out)
{
  const int C = 16, D = 41, H = 100, W = 88;
  const long tot = 2L * C * D * H * W;
  const long i = (long)blockIdx.x * blockDim.x + threadIdx.x;
  if (i >= tot) return;
  long t = i;
  const int w = (int)(t % W); t /= W;
  const int h = (int)(t % H); t /= H;
  const int d = (int)(t % D); t /= D;
  const int c = (int)(t % C);
  const int b = (int)(t / C);
  out[i] = u1[(((long)(b * D + d) * H + h) * W + w) * C + c];
}

// ---------------------------------------------------------------------------
// Host orchestration
// ---------------------------------------------------------------------------
struct Lvl { int D, H, W; };
static inline long sites(const Lvl& l) { return 2L * l.D * l.H * l.W; }

extern "C" void kernel_launch(void* const* d_in, const int* in_sizes, int n_in,
                              void* d_out, int out_size, void* d_ws, size_t ws_size,
                              hipStream_t stream)
{
  (void)in_sizes; (void)n_in; (void)out_size; (void)ws_size;

  const float* feats = (const float*)d_in[0];
  const int*   coors = (const int*)d_in[1];
  auto P = [&](int i) -> const float* { return (const float*)d_in[i]; };

  // Param leaf indices (pytree order: nested dict keys sorted; proj=None elided).
  // basic block (12): 0:b1 1:b2 2..5:bn1{b,g,m,v} 6..9:bn2{b,g,m,v} 10:w1 11:w2
  // down level (25): block1, block2, pool_w
  // up block: b1,b2,bn1(4),bn2(4),[proj],up_w,w1,w2
  const int C1 = 2, C2 = 27, C3 = 52, C4 = 77, CIN = 102;
  const int U1 = 107, U2 = 120, U3 = 134, U4 = 148;

  const Lvl L0{41, 100, 88}, L1{21, 50, 44}, L2{11, 25, 22},
            L3{5, 13, 11}, L4{2, 13, 11};
  const long s0 = sites(L0), s1 = sites(L1), s2 = sites(L2),
             s3 = sites(L3), s4 = sites(L4);

  // ---- bump allocator over d_ws (everything written before read) ----------
  char* wsbase = (char*)d_ws; size_t wsoff = 0;
  auto alloc = [&](size_t bytes) -> void* {
    void* ptr = wsbase + wsoff;
    wsoff += (bytes + 255) & ~(size_t)255;
    return ptr;
  };
  float* mask0 = (float*)alloc(s0 * 4);
  float* mask1 = (float*)alloc(s1 * 4);
  float* mask2 = (float*)alloc(s2 * 4);
  float* mask3 = (float*)alloc(s3 * 4);
  float* mask4 = (float*)alloc(s4 * 4);

  float* F0[3]; half_t* H0[3];
  for (int i = 0; i < 3; ++i) F0[i] = (float*)alloc(s0 * 16 * 4);
  for (int i = 0; i < 3; ++i) H0[i] = (half_t*)alloc(s0 * 16 * 2);
  float* F1[4]; half_t* H1[4];
  F1[0] = (float*)alloc(s1 * 16 * 4);  F1[1] = (float*)alloc(s1 * 16 * 4);
  F1[2] = (float*)alloc(s1 * 32 * 4);  F1[3] = (float*)alloc(s1 * 32 * 4);
  H1[0] = (half_t*)alloc(s1 * 16 * 2); H1[1] = (half_t*)alloc(s1 * 16 * 2);
  H1[2] = (half_t*)alloc(s1 * 32 * 2); H1[3] = (half_t*)alloc(s1 * 32 * 2);
  float* F2[4]; half_t* H2[4];
  F2[0] = (float*)alloc(s2 * 32 * 4);  F2[1] = (float*)alloc(s2 * 32 * 4);
  F2[2] = (float*)alloc(s2 * 64 * 4);  F2[3] = (float*)alloc(s2 * 64 * 4);
  H2[0] = (half_t*)alloc(s2 * 32 * 2); H2[1] = (half_t*)alloc(s2 * 32 * 2);
  H2[2] = (half_t*)alloc(s2 * 64 * 2); H2[3] = (half_t*)alloc(s2 * 64 * 2);
  float* F3[4]; half_t* H3[4];
  F3[0] = (float*)alloc(s3 * 64 * 4);   F3[1] = (float*)alloc(s3 * 64 * 4);
  F3[2] = (float*)alloc(s3 * 128 * 4);  F3[3] = (float*)alloc(s3 * 128 * 4);
  H3[0] = (half_t*)alloc(s3 * 64 * 2);  H3[1] = (half_t*)alloc(s3 * 64 * 2);
  H3[2] = (half_t*)alloc(s3 * 128 * 2); H3[3] = (half_t*)alloc(s3 * 128 * 2);
  float*  F4 = (float*)alloc(s4 * 128 * 4);
  half_t* H4 = (half_t*)alloc(s4 * 128 * 2);

  // ---- pack weights + dispatch templated conv -----------------------------
  auto conv_full = [&](Lvl li, int Cin, Lvl lo, int Cout,
                       int KD, int KH, int KW, int SD, int SH, int SW,
                       int PD, int PH, int PW, int LD, int LH, int LW,
                       int flip, int w_iodhw,
                       const half_t* in, const float* w, const float* bias,
                       const float* bnb, const float* bng,
                       const float* bnm, const float* bnv,
                       const float* mask, const float* resid,
                       float* outf, half_t* outh, int relu) {
    const int KVv    = KD * KH * KW;
    const int Ktot   = KVv * Cin;
    const int NT     = Cout / 16;
    const int kiters = (Ktot + 31) / 32;
    const int CL     = __builtin_ctz(Cin);
    long wsCo, wsCi;
    if (w_iodhw) { wsCi = (long)Cout * KVv; wsCo = KVv; }   // IODHW (transposed)
    else         { wsCo = (long)Cin  * KVv; wsCi = KVv; }   // OIDHW / OC
    const long wsKd = (long)KH * KW, wsKh = KW, wsKw = 1;
    const int totE = kiters * NT * 512;
    half_t* wpk = (half_t*)alloc((size_t)totE * 2);
    pack_weights_kernel<<<(totE + 255) / 256, 256, 0, stream>>>(
        w, wpk, Cin, Cout, KD, KH, KW, flip,
        wsCo, wsCi, wsKd, wsKh, wsKw, totE, Ktot);

    ConvP p{};
    p.ID = li.D; p.IH = li.H; p.IW = li.W;
    p.OD = lo.D; p.OH = lo.H; p.OW = lo.W;
    p.SD = SD; p.SH = SH; p.SW = SW;
    p.PD = PD; p.PH = PH; p.PW = PW;
    p.Mtot = (int)sites(lo);
    p.has_bn = (bng != nullptr) ? 1 : 0;
    p.do_relu = relu;
    const int blocks = (p.Mtot + 63) / 64;
    const int kcode = KD * 100 + KH * 10 + KW;
    bool done = false;
#define CONV_CASE(nt, kd, kh, kw, cl, ld, lh, lw)                              \
    if (!done && NT == nt && kcode == (kd * 100 + kh * 10 + kw) && CL == cl && \
        LD == ld && LH == lh && LW == lw) {                                    \
      conv3d_wmma_kernel<nt, kd, kh, kw, cl, ld, lh, lw>                       \
          <<<blocks, CONV_BLOCK, 0, stream>>>(                                 \
              p, in, wpk, bias, bnb, bng, bnm, bnv, mask, resid, outf, outh);  \
      done = true;                                                             \
    }
    CONV_CASE(1, 3, 3, 3, 4, 1, 1, 1)   // 16 -> 16 subm / pool1 / up1 w1,w2
    CONV_CASE(1, 3, 3, 3, 5, 2, 2, 2)   // up1 inv (32 -> 16)
    CONV_CASE(1, 3, 3, 3, 6, 2, 2, 2)   // up2 inv (64 -> 16)
    CONV_CASE(2, 3, 3, 3, 4, 1, 1, 1)   // pool2 / up2 w1 (16 -> 32)
    CONV_CASE(2, 3, 3, 3, 5, 1, 1, 1)   // 32 -> 32 subm / up2 w2
    CONV_CASE(2, 3, 3, 3, 7, 2, 2, 2)   // up3 inv (128 -> 32)
    CONV_CASE(4, 3, 3, 3, 5, 1, 1, 1)   // pool3 / up3 w1 (32 -> 64)
    CONV_CASE(4, 3, 3, 3, 6, 1, 1, 1)   // 64 -> 64 subm / up3 w2
    CONV_CASE(8, 3, 3, 3, 6, 1, 1, 1)   // up4 w1 (64 -> 128)
    CONV_CASE(8, 3, 3, 3, 7, 1, 1, 1)   // up4 w2 (128 -> 128)
    CONV_CASE(8, 3, 1, 1, 6, 1, 1, 1)   // pool4 (64 -> 128)
    CONV_CASE(4, 3, 1, 1, 7, 2, 1, 1)   // up4 inv (128 -> 64)
    CONV_CASE(8, 1, 1, 1, 6, 1, 1, 1)   // proj4 (64 -> 128)
    CONV_CASE(4, 1, 1, 1, 5, 1, 1, 1)   // proj3 (32 -> 64)
    CONV_CASE(2, 1, 1, 1, 4, 1, 1, 1)   // proj2 (16 -> 32)
#undef CONV_CASE
  };

  // ---- scatter voxels into dense grid + active mask -----------------------
  (void)hipMemsetAsync(F0[0], 0, s0 * 16 * 4, stream);
  (void)hipMemsetAsync(mask0, 0, s0 * 4, stream);
  scatter_kernel<<<(60000 + 255) / 256, 256, 0, stream>>>(feats, coors, F0[0], mask0, 60000);
  cvt_f16_kernel<<<(int)((s0 * 16 + 255) / 256), 256, 0, stream>>>(F0[0], H0[0], s0 * 16);

  // ---- mask pyramid (window-max of input mask) ----------------------------
  auto maxpool = [&](Lvl li, Lvl lo, int KD, int KH, int KW,
                     int SD, int SH, int SW, int PD, int PH, int PW,
                     const float* mi, float* mo) {
    maxpool_mask_kernel<<<(int)((sites(lo) + 255) / 256), 256, 0, stream>>>(
        mi, mo, li.D, li.H, li.W, lo.D, lo.H, lo.W,
        KD, KH, KW, SD, SH, SW, PD, PH, PW);
  };
  maxpool(L0, L1, 3,3,3, 2,2,2, 1,1,1, mask0, mask1);
  maxpool(L1, L2, 3,3,3, 2,2,2, 1,1,1, mask1, mask2);
  maxpool(L2, L3, 3,3,3, 2,2,2, 0,1,1, mask2, mask3);
  maxpool(L3, L4, 3,1,1, 2,1,1, 0,0,0, mask3, mask4);

  // ---- basic block: two subm convs, BN+ReLU, residual ---------------------
  auto run_block = [&](int bb, Lvl l, int C, const half_t* xh, const float* xf,
                       const float* mk, float* of, half_t* oh, half_t* th) {
    conv_full(l, C, l, C, 3,3,3, 1,1,1, 1,1,1, 1,1,1, 0, 0,
              xh, P(bb + 10), P(bb + 0),
              P(bb + 2), P(bb + 3), P(bb + 4), P(bb + 5),
              mk, nullptr, nullptr, th, 1);
    conv_full(l, C, l, C, 3,3,3, 1,1,1, 1,1,1, 1,1,1, 0, 0,
              th, P(bb + 11), P(bb + 1),
              P(bb + 6), P(bb + 7), P(bb + 8), P(bb + 9),
              mk, xf, of, oh, 1);
  };

  // ---- encoder ------------------------------------------------------------
  conv_full(L0, 16, L0, 16, 3,3,3, 1,1,1, 1,1,1, 1,1,1, 0, 0,
            H0[0], P(CIN + 4), nullptr,
            P(CIN + 0), P(CIN + 1), P(CIN + 2), P(CIN + 3),
            mask0, nullptr, F0[1], H0[1], 1);
  run_block(C1 + 0,  L0, 16, H0[1], F0[1], mask0, F0[0], H0[0], H0[2]);
  run_block(C1 + 12, L0, 16, H0[0], F0[0], mask0, F0[1], H0[1], H0[2]);  // x1
  conv_full(L0, 16, L1, 16, 3,3,3, 2,2,2, 1,1,1, 1,1,1, 0, 0,
            H0[1], P(C1 + 24), nullptr, nullptr, nullptr, nullptr, nullptr,
            mask1, nullptr, F1[0], H1[0], 0);                             // x1p

  run_block(C2 + 0,  L1, 16, H1[0], F1[0], mask1, F1[1], H1[1], H1[3]);
  run_block(C2 + 12, L1, 16, H1[1], F1[1], mask1, F1[0], H1[0], H1[3]);  // x2
  conv_full(L1, 16, L2, 32, 3,3,3, 2,2,2, 1,1,1, 1,1,1, 0, 0,
            H1[0], P(C2 + 24), nullptr, nullptr, nullptr, nullptr, nullptr,
            mask2, nullptr, F2[0], H2[0], 0);                             // x2p

  run_block(C3 + 0,  L2, 32, H2[0], F2[0], mask2, F2[1], H2[1], H2[3]);
  run_block(C3 + 12, L2, 32, H2[1], F2[1], mask2, F2[0], H2[0], H2[3]);  // x3
  conv_full(L2, 32, L3, 64, 3,3,3, 2,2,2, 0,1,1, 1,1,1, 0, 0,
            H2[0], P(C3 + 24), nullptr, nullptr, nullptr, nullptr, nullptr,
            mask3, nullptr, F3[0], H3[0], 0);                             // x3p

  run_block(C4 + 0,  L3, 64, H3[0], F3[0], mask3, F3[1], H3[1], H3[3]);
  run_block(C4 + 12, L3, 64, H3[1], F3[1], mask3, F3[0], H3[0], H3[3]);  // x4
  conv_full(L3, 64, L4, 128, 3,1,1, 2,1,1, 0,0,0, 1,1,1, 0, 0,
            H3[0], P(C4 + 24), nullptr, nullptr, nullptr, nullptr, nullptr,
            mask4, nullptr, F4, H4, 0);                                   // x4p

  // ---- decoder ------------------------------------------------------------
  // up4: inv (128->64, k(3,1,1), s(2,1,1), p0) + x4 skip; proj 64->128
  conv_full(L4, 128, L3, 64, 3,1,1, 1,1,1, 2,0,0, 2,1,1, 1, 1,
            H4, P(U4 + 11), nullptr, nullptr, nullptr, nullptr, nullptr,
            mask3, F3[0], F3[1], H3[1], 0);                               // merged4
  conv_full(L3, 64, L3, 128, 1,1,1, 1,1,1, 0,0,0, 1,1,1, 0, 0,
            H3[1], P(U4 + 10), nullptr, nullptr, nullptr, nullptr, nullptr,
            nullptr, nullptr, F3[2], nullptr, 0);                         // identity4
  conv_full(L3, 64, L3, 128, 3,3,3, 1,1,1, 1,1,1, 1,1,1, 0, 0,
            H3[1], P(U4 + 12), P(U4 + 0),
            P(U4 + 2), P(U4 + 3), P(U4 + 4), P(U4 + 5),
            mask3, nullptr, nullptr, H3[2], 1);
  conv_full(L3, 128, L3, 128, 3,3,3, 1,1,1, 1,1,1, 1,1,1, 0, 0,
            H3[2], P(U4 + 13), P(U4 + 1),
            P(U4 + 6), P(U4 + 7), P(U4 + 8), P(U4 + 9),
            mask3, F3[2], F3[3], H3[3], 1);                               // u4

  // up3: inv (128->32, k3, s2, p(0,1,1)) + x3 skip; proj 32->64
  conv_full(L3, 128, L2, 32, 3,3,3, 1,1,1, 2,1,1, 2,2,2, 1, 1,
            H3[3], P(U3 + 11), nullptr, nullptr, nullptr, nullptr, nullptr,
            mask2, F2[0], F2[1], H2[1], 0);                               // merged3
  conv_full(L2, 32, L2, 64, 1,1,1, 1,1,1, 0,0,0, 1,1,1, 0, 0,
            H2[1], P(U3 + 10), nullptr, nullptr, nullptr, nullptr, nullptr,
            nullptr, nullptr, F2[2], nullptr, 0);                         // identity3
  conv_full(L2, 32, L2, 64, 3,3,3, 1,1,1, 1,1,1, 1,1,1, 0, 0,
            H2[1], P(U3 + 12), P(U3 + 0),
            P(U3 + 2), P(U3 + 3), P(U3 + 4), P(U3 + 5),
            mask2, nullptr, nullptr, H2[2], 1);
  conv_full(L2, 64, L2, 64, 3,3,3, 1,1,1, 1,1,1, 1,1,1, 0, 0,
            H2[2], P(U3 + 13), P(U3 + 1),
            P(U3 + 6), P(U3 + 7), P(U3 + 8), P(U3 + 9),
            mask2, F2[2], F2[3], H2[3], 1);                               // u3

  // up2: inv (64->16, k3, s2, p1) + x2 skip; proj 16->32
  conv_full(L2, 64, L1, 16, 3,3,3, 1,1,1, 1,1,1, 2,2,2, 1, 1,
            H2[3], P(U2 + 11), nullptr, nullptr, nullptr, nullptr, nullptr,
            mask1, F1[0], F1[1], H1[1], 0);                               // merged2
  conv_full(L1, 16, L1, 32, 1,1,1, 1,1,1, 0,0,0, 1,1,1, 0, 0,
            H1[1], P(U2 + 10), nullptr, nullptr, nullptr, nullptr, nullptr,
            nullptr, nullptr, F1[2], nullptr, 0);                         // identity2
  conv_full(L1, 16, L1, 32, 3,3,3, 1,1,1, 1,1,1, 1,1,1, 0, 0,
            H1[1], P(U2 + 12), P(U2 + 0),
            P(U2 + 2), P(U2 + 3), P(U2 + 4), P(U2 + 5),
            mask1, nullptr, nullptr, H1[2], 1);
  conv_full(L1, 32, L1, 32, 3,3,3, 1,1,1, 1,1,1, 1,1,1, 0, 0,
            H1[2], P(U2 + 13), P(U2 + 1),
            P(U2 + 6), P(U2 + 7), P(U2 + 8), P(U2 + 9),
            mask1, F1[2], F1[3], H1[3], 1);                               // u2

  // up1: inv (32->16, k3, s2, p1) + x1 skip; no proj (identity = merged)
  conv_full(L1, 32, L0, 16, 3,3,3, 1,1,1, 1,1,1, 2,2,2, 1, 1,
            H1[3], P(U1 + 10), nullptr, nullptr, nullptr, nullptr, nullptr,
            mask0, F0[1], F0[0], H0[0], 0);                               // merged1
  conv_full(L0, 16, L0, 16, 3,3,3, 1,1,1, 1,1,1, 1,1,1, 0, 0,
            H0[0], P(U1 + 11), P(U1 + 0),
            P(U1 + 2), P(U1 + 3), P(U1 + 4), P(U1 + 5),
            mask0, nullptr, nullptr, H0[2], 1);
  conv_full(L0, 16, L0, 16, 3,3,3, 1,1,1, 1,1,1, 1,1,1, 0, 0,
            H0[2], P(U1 + 12), P(U1 + 1),
            P(U1 + 6), P(U1 + 7), P(U1 + 8), P(U1 + 9),
            mask0, F0[0], F0[2], nullptr, 1);                             // u1

  // ---- outputs: (ret_bev, u1) concatenated flat ---------------------------
  float* out = (float*)d_out;
  bev_out_kernel<<<(73216 + 255) / 256, 256, 0, stream>>>(F4, out);
  u1_out_kernel<<<(int)((s0 * 16 + 255) / 256), 256, 0, stream>>>(F0[2], out + 73216);
}
